// TDT_Interaction_5025111736707
// MI455X (gfx1250) — matrix operand order repaired and checked by reading, not compiled, not run on hardware
//
#include <hip/hip_runtime.h>
#include <hip/hip_bf16.h>

typedef __attribute__((ext_vector_type(16))) _Float16 v16h;
typedef __attribute__((ext_vector_type(8)))  float    v8f;

constexpr int Bn   = 8;
constexpr int An   = 512;
constexpr int NBHn = 64;
constexpr int Fn   = 128;
constexpr int Gn   = 50;
constexpr int Hn   = 8;
constexpr int DHn  = 16;
constexpr float CUT = 5.0f;

// ---- WMMA fragment loaders (CDNA5 §7.12.2 layouts, wave32) ----
// A (16x32 f16): lane m=L%16; half h -> k = h + 8*(h/8) + 8*(L/16)
__device__ __forceinline__ v16h frag_a(const _Float16* base, int stride,
                                       int mbase, int lane, int kbase) {
  union { v16h v; unsigned u[8]; } f;
  const _Float16* row = base + (size_t)(mbase + (lane & 15)) * stride
                        + kbase + ((lane >> 4) & 1) * 8;
#pragma unroll
  for (int j = 0; j < 8; ++j) {
    int k0 = 2 * j + ((j >= 4) ? 8 : 0);
    f.u[j] = *(const unsigned*)(row + k0);
  }
  return f.v;
}

// B (32x16 f16), read from transposed storage BT[n][k]:
// lane n=L%16; half h -> k = h + 8*(h/8) + 8*(L/16)
__device__ __forceinline__ v16h frag_b(const _Float16* baseT, int stride,
                                       int nbase, int lane, int kbase) {
  union { v16h v; unsigned u[8]; } f;
  const _Float16* row = baseT + (size_t)(nbase + (lane & 15)) * stride
                        + kbase + ((lane >> 4) & 1) * 8;
#pragma unroll
  for (int j = 0; j < 8; ++j) f.u[j] = *(const unsigned*)(row + 2 * j + ((j >= 4) ? 8 : 0));
  return f.v;
}

// K / V projection: sKVB[m][col] = sum_k sNbhB[m][k] * Wsrc[k][col]
__device__ __forceinline__ void project_nbh(const float* __restrict__ Wsrc,
                                            _Float16* __restrict__ sWT,
                                            const _Float16* __restrict__ sNbhB,
                                            _Float16* __restrict__ sKVB,
                                            int tid, int lane, int wv) {
  const int mbase = wv * 16;
  const int hi8   = ((lane >> 4) & 1) * 8;
  const int fl    = lane & 15;
  for (int nh = 0; nh < 2; ++nh) {
    // stage transposed weight half: sWT[j][k] = W[k][nh*64+j]  (f32 -> f16)
    for (int idx = tid; idx < 64 * 128; idx += 128) {
      int j = idx & 63;
      int k = idx >> 6;
      sWT[j * 132 + k] = (_Float16)Wsrc[(size_t)k * Fn + nh * 64 + j];
    }
    __syncthreads();
#pragma unroll
    for (int nt = 0; nt < 4; ++nt) {
      v8f acc = {};
#pragma unroll
      for (int kt = 0; kt < 4; ++kt) {
        v16h a  = frag_a(sNbhB, 132, mbase, lane, kt * 32);
        v16h bm = frag_b(sWT,   132, nt * 16, lane, kt * 32);
        acc = __builtin_amdgcn_wmma_f32_16x16x32_f16(false, a, false, bm,
                                                     (short)0, acc, false, false);
      }
      int col = nh * 64 + nt * 16 + fl;
#pragma unroll
      for (int r = 0; r < 8; ++r)
        sKVB[(size_t)(mbase + hi8 + r) * 132 + col] = (_Float16)acc[r];
    }
    __syncthreads();
  }
}

__global__ __launch_bounds__(128) void egoattn_kernel(
    const float* __restrict__ x,     const float* __restrict__ r_ij,
    const float* __restrict__ f_ij,  const float* __restrict__ W_filt,
    const float* __restrict__ b_filt,const float* __restrict__ Wq,
    const float* __restrict__ Wk,    const float* __restrict__ Wv,
    const float* __restrict__ Wo,    const float* __restrict__ bo,
    const int* __restrict__ nbrs,    const unsigned char* __restrict__ nmask,
    float* __restrict__ out)
{
  __shared__ __align__(16) _Float16 sWT[8704];      // weights, transposed (f16)
  __shared__ __align__(16) _Float16 sNbh[64 * 132]; // filtered neighbor feats
  __shared__ __align__(16) _Float16 sKV[8448];      // f_ij, then K, then V
  __shared__ float sC[NBHn];
  __shared__ int   sNbr[NBHn];
  __shared__ float sXrow[Fn];
  __shared__ float sQ[Fn];
  __shared__ float sScores[Hn * NBHn];
  __shared__ float sMsg[Fn];

  const int tid  = threadIdx.x;
  const int lane = tid & 31;
  const int wv   = tid >> 5;
  const int ba   = blockIdx.x;          // b*A + a
  const int b    = ba >> 9;             // A = 512
  const size_t baseNb = (size_t)ba * NBHn;

  // ---- per-neighbor scalars: cosine cutoff, neighbor ids ----
  if (tid < NBHn) {
    float r = r_ij[baseNb + tid];
    float c = 0.5f * (cosf(3.14159265358979323846f * r / CUT) + 1.0f);
    sC[tid]   = (r < CUT) ? c : 0.0f;
    sNbr[tid] = nbrs[baseNb + tid];
  }
  sXrow[tid] = x[(size_t)ba * Fn + tid];

  // ---- stage f_ij (64x50 -> 64x64 f16, stride 68) into sKV region ----
  for (int idx = tid; idx < 64 * 64; idx += 128) {
    int n = idx >> 6, g = idx & 63;
    float v = (g < Gn) ? f_ij[(baseNb + n) * Gn + g] : 0.0f;
    sKV[n * 68 + g] = (_Float16)v;
  }
  // ---- stage W_filt transposed (f16): sWT[n][g] = W_filt[g][n] ----
  for (int idx = tid; idx < 128 * 64; idx += 128) {
    int g = idx >> 7, n = idx & 127;
    float v = (g < Gn) ? W_filt[(size_t)g * Fn + n] : 0.0f;
    sWT[n * 68 + g] = (_Float16)v;
  }
  __syncthreads();

  // ---- GEMM1: Wfilter = f_ij @ W_filt ; fuse bias, cutoff, gather, modulate ----
  {
    const int mbase = wv * 16;
    const int hi8   = ((lane >> 4) & 1) * 8;
    const int fl    = lane & 15;
#pragma unroll
    for (int nt = 0; nt < 8; ++nt) {
      v8f acc = {};
#pragma unroll
      for (int kt = 0; kt < 2; ++kt) {
        v16h a  = frag_a(sKV, 68, mbase, lane, kt * 32);
        v16h bm = frag_b(sWT, 68, nt * 16, lane, kt * 32);
        acc = __builtin_amdgcn_wmma_f32_16x16x32_f16(false, a, false, bm,
                                                     (short)0, acc, false, false);
      }
      int fcol  = nt * 16 + fl;
      float bfv = b_filt[fcol];
#pragma unroll
      for (int r = 0; r < 8; ++r) {
        int m = mbase + hi8 + r;                           // neighbor row
        float wval = (acc[r] + bfv) * sC[m];               // filter * cutoff
        float xg = x[((size_t)b * An + sNbr[m]) * Fn + fcol]; // gather
        sNbh[(size_t)m * 132 + fcol] = (_Float16)(wval * xg);
      }
    }
  }

  // ---- q = x_row @ Wq, scaled by 1/sqrt(DH) (fp32) ----
  {
    float acc = 0.0f;
    for (int k = 0; k < Fn; ++k) acc += sXrow[k] * Wq[(size_t)k * Fn + tid];
    sQ[tid] = acc * 0.25f;
  }
  __syncthreads();

  // ---- K = nbh @ Wk (f16 WMMA, f32 accum) -> sKV ----
  project_nbh(Wk, sWT, sNbh, sKV, tid, lane, wv);

  // ---- scores + mask (fp32) ----
  for (int p = tid; p < Hn * NBHn; p += 128) {
    int h = p >> 6, n = p & 63;
    float s = 0.0f;
#pragma unroll
    for (int d = 0; d < DHn; ++d)
      s += sQ[h * 16 + d] * (float)sKV[n * 132 + h * 16 + d];
    bool mk = nmask[baseNb + n] != 0;
    sScores[h * 64 + n] = mk ? s : -1e9f;
  }
  __syncthreads();

  // ---- softmax over neighbors, one thread per head (fp32) ----
  if (tid < Hn) {
    float mx = -3.4e38f;
    for (int n = 0; n < NBHn; ++n) mx = fmaxf(mx, sScores[tid * 64 + n]);
    float sum = 0.0f;
    for (int n = 0; n < NBHn; ++n) {
      float e2 = expf(sScores[tid * 64 + n] - mx);
      sScores[tid * 64 + n] = e2;
      sum += e2;
    }
    float inv = 1.0f / sum;
    for (int n = 0; n < NBHn; ++n) sScores[tid * 64 + n] *= inv;
  }
  __syncthreads();

  // ---- V = nbh @ Wv (overwrites K in sKV; scores already consumed) ----
  project_nbh(Wv, sWT, sNbh, sKV, tid, lane, wv);

  // ---- msg[h,d] = sum_n attn[h,n] * v[n,h*16+d] ----
  {
    int h = tid >> 4;
    float acc = 0.0f;
    for (int n = 0; n < NBHn; ++n)
      acc += sScores[h * 64 + n] * (float)sKV[n * 132 + tid];
    sMsg[tid] = acc;
  }
  __syncthreads();

  // ---- v_out = msg @ Wo + bo ; residual ----
  {
    float o = bo[tid];
    for (int d = 0; d < Fn; ++d) o += sMsg[d] * Wo[(size_t)d * Fn + tid];
    out[(size_t)ba * Fn + tid] = sXrow[tid] + o;
  }
}

extern "C" void kernel_launch(void* const* d_in, const int* in_sizes, int n_in,
                              void* d_out, int out_size, void* d_ws, size_t ws_size,
                              hipStream_t stream) {
  (void)in_sizes; (void)n_in; (void)out_size; (void)d_ws; (void)ws_size;
  // setup_inputs order: e, x, t, r_ij, f_ij, W_filt, b_filt, Wq, Wk, Wv, Wo, bo,
  //                     neighbors, neighbor_mask   (e and t are unused)
  const float* x      = (const float*)d_in[1];
  const float* r_ij   = (const float*)d_in[3];
  const float* f_ij   = (const float*)d_in[4];
  const float* W_filt = (const float*)d_in[5];
  const float* b_filt = (const float*)d_in[6];
  const float* Wq     = (const float*)d_in[7];
  const float* Wk     = (const float*)d_in[8];
  const float* Wv     = (const float*)d_in[9];
  const float* Wo     = (const float*)d_in[10];
  const float* bo     = (const float*)d_in[11];
  const int*   nbrs   = (const int*)d_in[12];
  const unsigned char* nmask = (const unsigned char*)d_in[13];
  float* out = (float*)d_out;

  egoattn_kernel<<<Bn * An, 128, 0, stream>>>(x, r_ij, f_ij, W_filt, b_filt,
                                              Wq, Wk, Wv, Wo, bo, nbrs, nmask, out);
}
